// BF16SafeAttention_44195213476315
// MI455X (gfx1250) — matrix operand order repaired and checked by reading, not compiled, not run on hardware
//
#include <hip/hip_runtime.h>

// ---------------------------------------------------------------------------
// Shapes (fixed by the reference)
// ---------------------------------------------------------------------------
#define D_MODEL 1024
#define NHEAD   16
#define HEAD_DIM 64
#define BATCH   2
#define SQ      2048
#define SK      2048
#define M_ROWS  (BATCH * SQ)          // 4096

typedef unsigned short u16;
typedef unsigned int   u32;
typedef __attribute__((ext_vector_type(16))) __bf16 v16bf;
typedef __attribute__((ext_vector_type(8)))  float  v8f;

union U16V {
    v16bf v;
    uint4 q[2];
};

// float -> bf16 round-to-nearest-even (bit pattern)
__device__ __forceinline__ u16 f2bf(float f) {
    u32 u = __float_as_uint(f);
    u += 0x7fffu + ((u >> 16) & 1u);
    return (u16)(u >> 16);
}

// ---------------------------------------------------------------------------
// WMMA fragment loaders (layouts per CDNA5 ISA §7.12.2, wave32)
//
// A (16x32, MxK): lane L -> row M=L%16; 16B chunks at k0+(L/16)*8 and
//                 k0+16+(L/16)*8  (VGPR0-3 / VGPR4-7 halves)
// B (32x16, KxN): lane L -> col N=L%16; 16B chunks at k0+(L/16)*8 and
//                 k0+16+(L/16)*8  (same k order as A)
// C/D (16x16 f32): lane = N + 16*(M/8), vgpr = M%8
// ---------------------------------------------------------------------------
__device__ __forceinline__ v16bf load_a16(const u16* base, int ldk, int row,
                                          int k0, int half) {
    const uint4* p =
        reinterpret_cast<const uint4*>(base + (size_t)row * ldk + k0 + half * 8);
    U16V u;
    u.q[0] = p[0];   // k0 + half*8      .. +7
    u.q[1] = p[2];   // k0 + 16 + half*8 .. +7
    return u.v;
}

__device__ __forceinline__ v16bf load_b16(const u16* base, int ldk, int row,
                                          int k0, int half) {
    const uint4* p =
        reinterpret_cast<const uint4*>(base + (size_t)row * ldk + k0 + half * 8);
    U16V u;
    u.q[0] = p[0];
    u.q[1] = p[2];
    return u.v;
}

__device__ __forceinline__ v8f wmma_bf(v16bf a, v16bf b, v8f c) {
    return __builtin_amdgcn_wmma_f32_16x16x32_bf16(
        /*neg_a=*/false, a, /*neg_b=*/false, b,
        /*c_mod=*/(short)0, c, /*reuse_a=*/false, /*reuse_b=*/false);
}

__device__ __forceinline__ float rmax16(float x) {
    x = fmaxf(x, __shfl_xor(x, 1, 16));
    x = fmaxf(x, __shfl_xor(x, 2, 16));
    x = fmaxf(x, __shfl_xor(x, 4, 16));
    x = fmaxf(x, __shfl_xor(x, 8, 16));
    return x;
}
__device__ __forceinline__ float rsum16(float x) {
    x += __shfl_xor(x, 1, 16);
    x += __shfl_xor(x, 2, 16);
    x += __shfl_xor(x, 4, 16);
    x += __shfl_xor(x, 8, 16);
    return x;
}

// ---------------------------------------------------------------------------
// f32 -> bf16 bulk convert (vectorized: float4 in, 4x bf16 packed out)
// ---------------------------------------------------------------------------
__global__ void cvt_f32_bf16(const float* __restrict__ in, u16* __restrict__ out,
                             int n4) {
    int i = blockIdx.x * blockDim.x + threadIdx.x;
    if (i < n4) {
        float4 f = reinterpret_cast<const float4*>(in)[i];
        u32 lo = (u32)f2bf(f.x) | ((u32)f2bf(f.y) << 16);
        u32 hi = (u32)f2bf(f.z) | ((u32)f2bf(f.w) << 16);
        reinterpret_cast<uint2*>(out)[i] = make_uint2(lo, hi);
    }
}

// ---------------------------------------------------------------------------
// GEMM: C[M x N] = A[M x K] * W[N x K]^T + bias  (K = N = 1024, M = 4096)
// Each wave computes a 32x32 tile (4 WMMA per k-step, 2 b128 loads per WMMA).
// Block = 128 threads (4 waves) -> 64x64 tile. Grid = (64, 16).
// mode 0: store bf16 into [B,H,S,Dh]      (Q/K heads)
// mode 1: store bf16 into [B,H,Dh,S]      (V transposed, for PV B-operand)
// mode 2: store f32 row-major [M x N]     (final output)
// ---------------------------------------------------------------------------
__global__ __launch_bounds__(128) void gemm_bf16_wmma(
    const u16* __restrict__ A, const u16* __restrict__ W,
    const float* __restrict__ bias, void* __restrict__ outp, int mode) {
    const int lane = threadIdx.x & 31;
    const int wv   = threadIdx.x >> 5;
    const int half = lane >> 4;
    const int col  = lane & 15;
    const int mBase = blockIdx.x * 64 + (wv >> 1) * 32;
    const int nBase = blockIdx.y * 64 + (wv & 1) * 32;

    v8f acc00 = {}, acc01 = {}, acc10 = {}, acc11 = {};
    for (int k0 = 0; k0 < D_MODEL; k0 += 32) {
        if (k0 + 64 < D_MODEL) {  // pull next-next K-slab toward the WGP
            __builtin_prefetch(A + (size_t)(mBase + col) * D_MODEL + k0 + 64, 0, 1);
            __builtin_prefetch(W + (size_t)(nBase + col) * D_MODEL + k0 + 64, 0, 1);
        }
        v16bf a0 = load_a16(A, D_MODEL, mBase + col,      k0, half);
        v16bf a1 = load_a16(A, D_MODEL, mBase + 16 + col, k0, half);
        v16bf b0 = load_b16(W, D_MODEL, nBase + col,      k0, half);
        v16bf b1 = load_b16(W, D_MODEL, nBase + 16 + col, k0, half);
        acc00 = wmma_bf(a0, b0, acc00);
        acc01 = wmma_bf(a0, b1, acc01);
        acc10 = wmma_bf(a1, b0, acc10);
        acc11 = wmma_bf(a1, b1, acc11);
    }

    const float bv0 = bias[nBase + col];
    const float bv1 = bias[nBase + 16 + col];
    u16*   o16 = (u16*)outp;
    float* o32 = (float*)outp;

    #pragma unroll
    for (int i = 0; i < 2; i++) {
        #pragma unroll
        for (int j = 0; j < 2; j++) {
            v8f acc = i ? (j ? acc11 : acc10) : (j ? acc01 : acc00);
            const int mt = mBase + i * 16;
            const int nt = nBase + j * 16;
            const float bv = j ? bv1 : bv0;
            #pragma unroll
            for (int r = 0; r < 8; r++) {
                const int m = mt + r + half * 8;
                const int n = nt + col;
                const float vres = acc[r] + bv;
                if (mode == 2) {
                    o32[(size_t)m * D_MODEL + n] = vres;
                } else {
                    const int b = m >> 11, s = m & (SQ - 1);
                    const int h = n >> 6,  dh = n & (HEAD_DIM - 1);
                    if (mode == 0)
                        o16[(((size_t)(b * NHEAD + h)) * SQ + s) * HEAD_DIM + dh] =
                            f2bf(vres);
                    else
                        o16[(((size_t)(b * NHEAD + h)) * HEAD_DIM + dh) * SK + s] =
                            f2bf(vres);
                }
            }
        }
    }
}

// ---------------------------------------------------------------------------
// Flash attention, causal. One wave per (b, h, 16-query-row block).
// Qh, Kh: bf16 [B,H,S,64]; Vt: bf16 [B,H,64,S] (pre-transposed).
// Online softmax; P relayout C->A via a 1KB wave-private LDS tile (per-wave
// DS ops are in-order, so no barrier needed). Causal bound skips key tiles
// entirely above the diagonal.
// ---------------------------------------------------------------------------
__global__ __launch_bounds__(32) void flash_attn_bf16(
    const u16* __restrict__ Qh, const u16* __restrict__ Kh,
    const u16* __restrict__ Vt, u16* __restrict__ Oo) {
    __shared__ __align__(16) u16 P[16][32];

    const int lane = threadIdx.x;
    const int half = lane >> 4;
    const int col  = lane & 15;
    const int qTile = blockIdx.x & (SQ / 16 - 1);
    const int bh    = blockIdx.x >> 7;          // SQ/16 == 128
    const int qBase = qTile * 16;

    const u16* Qp = Qh + (size_t)bh * SQ * HEAD_DIM;
    const u16* Kp = Kh + (size_t)bh * SK * HEAD_DIM;
    const u16* Vp = Vt + (size_t)bh * HEAD_DIM * SK;

    const v16bf qa0 = load_a16(Qp, HEAD_DIM, qBase + col, 0,  half);
    const v16bf qa1 = load_a16(Qp, HEAD_DIM, qBase + col, 32, half);

    v8f o0 = {}, o1 = {}, o2 = {}, o3 = {};
    float mrow[8], lrow[8];
    #pragma unroll
    for (int r = 0; r < 8; r++) { mrow[r] = -INFINITY; lrow[r] = 0.f; }

    const float scale = 0.125f;          // 64^-0.5
    const int ktEnd = (qBase + 15) >> 5; // last 32-key tile touching diagonal

    for (int kt = 0; kt <= ktEnd; kt++) {
        const int kb = kt * 32;
        float s0[8], s1[8];

        {   // keys kb .. kb+15
            v16bf k0 = load_b16(Kp, HEAD_DIM, kb + col, 0,  half);
            v16bf k1 = load_b16(Kp, HEAD_DIM, kb + col, 32, half);
            v8f z = {};
            v8f acc = wmma_bf(qa0, k0, z);
            acc = wmma_bf(qa1, k1, acc);
            #pragma unroll
            for (int r = 0; r < 8; r++) {
                const int qi = qBase + r + half * 8;
                const int kj = kb + col;
                s0[r] = (kj <= qi) ? acc[r] * scale : -INFINITY;
            }
        }
        {   // keys kb+16 .. kb+31
            v16bf k0 = load_b16(Kp, HEAD_DIM, kb + 16 + col, 0,  half);
            v16bf k1 = load_b16(Kp, HEAD_DIM, kb + 16 + col, 32, half);
            v8f z = {};
            v8f acc = wmma_bf(qa0, k0, z);
            acc = wmma_bf(qa1, k1, acc);
            #pragma unroll
            for (int r = 0; r < 8; r++) {
                const int qi = qBase + r + half * 8;
                const int kj = kb + 16 + col;
                s1[r] = (kj <= qi) ? acc[r] * scale : -INFINITY;
            }
        }

        float alpha[8];
        #pragma unroll
        for (int r = 0; r < 8; r++) {
            float tm = rmax16(fmaxf(s0[r], s1[r]));
            float mnew = fmaxf(mrow[r], tm);
            alpha[r] = __expf(mrow[r] - mnew);
            mrow[r] = mnew;
        }
        #pragma unroll
        for (int r = 0; r < 8; r++) {
            s0[r] = __expf(s0[r] - mrow[r]);
            s1[r] = __expf(s1[r] - mrow[r]);
            lrow[r] = lrow[r] * alpha[r] + rsum16(s0[r] + s1[r]);
        }
        #pragma unroll
        for (int r = 0; r < 8; r++) {
            o0[r] *= alpha[r]; o1[r] *= alpha[r];
            o2[r] *= alpha[r]; o3[r] *= alpha[r];
        }

        // C-layout -> row-major P tile in LDS (bf16), then reload as A-frag
        #pragma unroll
        for (int r = 0; r < 8; r++) {
            const int pm = r + half * 8;
            P[pm][col]      = f2bf(s0[r]);
            P[pm][16 + col] = f2bf(s1[r]);
        }
        v16bf pa = load_a16(&P[0][0], 32, col, 0, half);

        // O += P[16x32] x V[32keys x 64dh]  (Vt rows = dh, keys contiguous)
        o0 = wmma_bf(pa, load_b16(Vp, SK,  0 + col, kb, half), o0);
        o1 = wmma_bf(pa, load_b16(Vp, SK, 16 + col, kb, half), o1);
        o2 = wmma_bf(pa, load_b16(Vp, SK, 32 + col, kb, half), o2);
        o3 = wmma_bf(pa, load_b16(Vp, SK, 48 + col, kb, half), o3);
    }

    // normalize and store bf16 into [B*S, D_MODEL] for the output projection
    const int b = bh >> 4, h = bh & 15;
    #pragma unroll
    for (int r = 0; r < 8; r++) {
        const float inv = 1.f / lrow[r];
        const int m = qBase + r + half * 8;
        const size_t rowoff = ((size_t)(b * SQ + m)) * D_MODEL + h * HEAD_DIM;
        Oo[rowoff +  0 + col] = f2bf(o0[r] * inv);
        Oo[rowoff + 16 + col] = f2bf(o1[r] * inv);
        Oo[rowoff + 32 + col] = f2bf(o2[r] * inv);
        Oo[rowoff + 48 + col] = f2bf(o3[r] * inv);
    }
}

// ---------------------------------------------------------------------------
// Host side
// ---------------------------------------------------------------------------
extern "C" void kernel_launch(void* const* d_in, const int* in_sizes, int n_in,
                              void* d_out, int out_size, void* d_ws,
                              size_t ws_size, hipStream_t stream) {
    const float* q  = (const float*)d_in[0];
    const float* k  = (const float*)d_in[1];
    const float* v  = (const float*)d_in[2];
    // d_in[3] = causal_mask (bool) -- causality computed analytically
    const float* Wq = (const float*)d_in[4];
    const float* bq = (const float*)d_in[5];
    const float* Wk = (const float*)d_in[6];
    const float* bk = (const float*)d_in[7];
    const float* Wv = (const float*)d_in[8];
    const float* bv = (const float*)d_in[9];
    const float* Wo = (const float*)d_in[10];
    const float* bo = (const float*)d_in[11];

    // Workspace layout (bf16 elements); total 32 Mi elems = 64 MiB
    const size_t ME = (size_t)1024 * 1024;
    u16* ws   = (u16*)d_ws;
    u16* Wq_b = ws;            // 1M
    u16* Wk_b = ws + 1 * ME;
    u16* Wv_b = ws + 2 * ME;
    u16* Wo_b = ws + 3 * ME;
    u16* q_b  = ws + 4 * ME;   // 4M each
    u16* k_b  = ws + 8 * ME;
    u16* v_b  = ws + 12 * ME;
    u16* Qh   = ws + 16 * ME;  // [B,H,S,Dh]
    u16* Kh   = ws + 20 * ME;  // [B,H,S,Dh]
    u16* Vt   = ws + 24 * ME;  // [B,H,Dh,S]
    u16* Ao   = ws + 28 * ME;  // [B*S, D]

    // f32 -> bf16 conversions (vectorized by 4)
    const int wN4 = (int)(ME / 4);          // weights: 1M elems
    const int aN4 = (int)(4 * ME / 4);      // activations: 4M elems
    cvt_f32_bf16<<<(wN4 + 255) / 256, 256, 0, stream>>>(Wq, Wq_b, wN4);
    cvt_f32_bf16<<<(wN4 + 255) / 256, 256, 0, stream>>>(Wk, Wk_b, wN4);
    cvt_f32_bf16<<<(wN4 + 255) / 256, 256, 0, stream>>>(Wv, Wv_b, wN4);
    cvt_f32_bf16<<<(wN4 + 255) / 256, 256, 0, stream>>>(Wo, Wo_b, wN4);
    cvt_f32_bf16<<<(aN4 + 255) / 256, 256, 0, stream>>>(q, q_b, aN4);
    cvt_f32_bf16<<<(aN4 + 255) / 256, 256, 0, stream>>>(k, k_b, aN4);
    cvt_f32_bf16<<<(aN4 + 255) / 256, 256, 0, stream>>>(v, v_b, aN4);

    // Projections
    dim3 gg(M_ROWS / 64, D_MODEL / 64);     // (64, 16)
    dim3 gb(128);
    gemm_bf16_wmma<<<gg, gb, 0, stream>>>(q_b, Wq_b, bq, Qh, 0);
    gemm_bf16_wmma<<<gg, gb, 0, stream>>>(k_b, Wk_b, bk, Kh, 0);
    gemm_bf16_wmma<<<gg, gb, 0, stream>>>(v_b, Wv_b, bv, Vt, 1);

    // Causal flash attention: one wave per 16 query rows per (b,h)
    flash_attn_bf16<<<dim3(BATCH * NHEAD * (SQ / 16)), 32, 0, stream>>>(
        Qh, Kh, Vt, Ao);

    // Output projection -> f32 d_out [B, SQ, D]
    gemm_bf16_wmma<<<gg, gb, 0, stream>>>(Ao, Wo_b, bo, d_out, 2);
}
